// SparseMoeBlock_51539607552742
// MI455X (gfx1250) — matrix operand order repaired and checked by reading, not compiled, not run on hardware
//
#include <hip/hip_runtime.h>
#include <math.h>

#define T_TOK 1024
#define H_DIM 1024
#define I_DIM 2816
#define E_EXP 16
#define TOPK 4

typedef __attribute__((ext_vector_type(16))) __bf16 v16bf;
typedef __attribute__((ext_vector_type(8))) float v8f;

// ---- f32 -> bf16 conversions ----

// RNE rounding add on f32 bits (used for activations)
__device__ __forceinline__ unsigned int bfround(unsigned int u) {
  return u + 0x7FFFu + ((u >> 16) & 1u);
}

__device__ __forceinline__ unsigned short f2bf(float f) {
  union { float f; unsigned int u; } c; c.f = f;
  return (unsigned short)(bfround(c.u) >> 16);
}

// activation pack: RNE, 2 add + 1 perm per pair
__device__ __forceinline__ unsigned int pack2bf_rne(float f0, float f1) {
  union { float f; unsigned int u; } c0, c1;
  c0.f = f0; c1.f = f1;
  return __builtin_amdgcn_perm(bfround(c1.u), bfround(c0.u), 0x07060302u);
}

// weight pack: 1 instruction per pair.
#if __has_builtin(__builtin_amdgcn_cvt_pk_bf16_f32)
typedef __attribute__((ext_vector_type(2))) __bf16 v2bf;
__device__ __forceinline__ unsigned int pack2bf_w(float f0, float f1) {
  union { v2bf v; unsigned int u; } c;
  c.v = __builtin_amdgcn_cvt_pk_bf16_f32(f0, f1);   // RNE, 1 op
  return c.u;
}
#else
__device__ __forceinline__ unsigned int pack2bf_w(float f0, float f1) {
  union { float f; unsigned int u; } c0, c1;
  c0.f = f0; c1.f = f1;
  return __builtin_amdgcn_perm(c1.u, c0.u, 0x07060302u);  // truncation, 1 op
}
#endif

// A-matrix 16x32 bf16 fragment from LDS (ISA 7.12.2 layout):
// lanes 0-15: M=lane, K = kb+0..7 and kb+16..23 ; lanes 16-31: M=lane-16, K = kb+8..15, kb+24..31
// 8+8 contiguous bf16 -> two ds_load_b128
__device__ __forceinline__ v16bf load_a_lds(const unsigned short* base, int ldk, int kb, int lane) {
  union { v16bf v; uint4 q[2]; } a;
  const int m = lane & 15;
  const int ko = (lane < 16) ? 0 : 8;
  const unsigned short* p = base + m * ldk + kb + ko;   // 16B-aligned
  a.q[0] = *(const uint4*)(p);
  a.q[1] = *(const uint4*)(p + 16);
  return a.v;
}

// B-matrix 32x16 bf16 fragment from global f32 (row-major KxN, leading dim ldw):
// lanes 0-15: N=lane, K=kb+0..7,kb+16..23 ; lanes 16-31: N=lane-16, K=kb+8..15,kb+24..31
// two K-rows packed per dword, 1 pack op per pair
__device__ __forceinline__ v16bf load_b_global(const float* W, int ldw, int kb, int nb, int lane) {
  union { v16bf v; unsigned int u[8]; } b;
  const int n = nb + (lane & 15);
  const int r = kb + ((lane < 16) ? 0 : 8);
  const float* p = W + (size_t)r * ldw + n;
#pragma unroll
  for (int j = 0; j < 8; ++j) {
    float f0 = p[0];
    float f1 = p[(size_t)ldw];
    p += (j == 3 ? 10 : 2) * (size_t)ldw;
    b.u[j] = pack2bf_w(f0, f1);
  }
  return b.v;
}

__device__ __forceinline__ v8f wmma_bf16(v16bf a, v16bf b, v8f c) {
  return __builtin_amdgcn_wmma_f32_16x16x32_bf16(false, a, false, b, (short)0, c, false, false);
}

__device__ __forceinline__ float fast_silu_mul(float g, float u) {
  // g * u / (1 + exp(-g)) with v_rcp_f32 instead of IEEE divide
  return g * u * __builtin_amdgcn_rcpf(1.f + __expf(-g));
}

// ---------------- kernel 1: zero output + expert counters ----------------
__global__ void zero_kernel(float* __restrict__ out, int* __restrict__ cnt) {
  size_t i = (size_t)blockIdx.x * blockDim.x + threadIdx.x;
  if (i < (size_t)T_TOK * H_DIM) out[i] = 0.f;
  if (i < E_EXP) cnt[i] = 0;
}

// ---------------- kernel 2: router (one wave per token) ----------------
__global__ void router_kernel(const float* __restrict__ x, const float* __restrict__ Wg,
                              float* __restrict__ combine) {
  const int tid = threadIdx.x;
  const int lane = tid & 31;
  const int wave = tid >> 5;
  const int t = blockIdx.x * 4 + wave;
  if (t >= T_TOK) return;

  float acc[E_EXP];
#pragma unroll
  for (int e = 0; e < E_EXP; ++e) acc[e] = 0.f;
  for (int h = lane; h < H_DIM; h += 32) {
    float xv = x[(size_t)t * H_DIM + h];
#pragma unroll
    for (int e = 0; e < E_EXP; ++e) acc[e] += xv * Wg[(size_t)e * H_DIM + h];
  }
#pragma unroll
  for (int e = 0; e < E_EXP; ++e) {
#pragma unroll
    for (int off = 16; off > 0; off >>= 1)
      acc[e] += __shfl_xor(acc[e], off, 32);
  }
  if (lane == 0) {
    float mx = acc[0];
#pragma unroll
    for (int e = 1; e < E_EXP; ++e) mx = fmaxf(mx, acc[e]);
    float p[E_EXP]; float s = 0.f;
#pragma unroll
    for (int e = 0; e < E_EXP; ++e) { p[e] = __expf(acc[e] - mx); s += p[e]; }
    float inv = 1.f / s;
#pragma unroll
    for (int e = 0; e < E_EXP; ++e) p[e] *= inv;

    float w[TOPK]; int id[TOPK]; float sum4 = 0.f;
    bool used[E_EXP];
#pragma unroll
    for (int e = 0; e < E_EXP; ++e) used[e] = false;
    for (int k = 0; k < TOPK; ++k) {
      int best = 0; float bv = -1.f;
      for (int e = 0; e < E_EXP; ++e)
        if (!used[e] && p[e] > bv) { bv = p[e]; best = e; }
      used[best] = true; id[k] = best; w[k] = bv; sum4 += bv;
    }
    float invs = 1.f / sum4;
    float row[E_EXP];
#pragma unroll
    for (int e = 0; e < E_EXP; ++e) row[e] = 0.f;
    for (int k = 0; k < TOPK; ++k) row[id[k]] = w[k] * invs;
#pragma unroll
    for (int e = 0; e < E_EXP; ++e) combine[(size_t)t * E_EXP + e] = row[e];
  }
}

// ---------------- kernel 3: build per-expert token lists ----------------
__global__ void scatter_kernel(const float* __restrict__ combine, int* __restrict__ cnt,
                               int* __restrict__ list) {
  int t = blockIdx.x * blockDim.x + threadIdx.x;
  if (t >= T_TOK) return;
#pragma unroll
  for (int e = 0; e < E_EXP; ++e) {
    if (combine[(size_t)t * E_EXP + e] > 0.f) {
      int pos = atomicAdd(&cnt[e], 1);
      list[(size_t)e * T_TOK + pos] = t;
    }
  }
}

// ---------------- kernel 4: fused SwiGLU FFN per (expert, 16-token tile) ----------------
__global__ void __launch_bounds__(256, 1)
moe_ffn_kernel(const float* __restrict__ x,
               const float* __restrict__ gate_w,
               const float* __restrict__ up_w,
               const float* __restrict__ down_w,
               const float* __restrict__ combine,
               const int* __restrict__ cnt,
               const int* __restrict__ list,
               float* __restrict__ out) {
  __shared__ __align__(16) unsigned short xs[16 * H_DIM];   // 32 KB : x tile, bf16
  __shared__ __align__(16) unsigned short hs[16 * 256];     // 8 KB  : h chunk, bf16
  __shared__ int toks[16];
  __shared__ float wts[16];

  const int e = blockIdx.y;
  const int tile = blockIdx.x;
  const int cnt_e = cnt[e];
  if (tile * 16 >= cnt_e) return;                 // uniform early-exit
  const int nvalid = min(16, cnt_e - tile * 16);

  const int tid = threadIdx.x;
  const int lane = tid & 31;
  const int wave = tid >> 5;                      // 8 waves

  if (tid < 16) {
    int m = tid;
    int tok = (m < nvalid) ? list[(size_t)e * T_TOK + tile * 16 + m] : 0;
    toks[m] = tok;
    wts[m] = (m < nvalid) ? combine[(size_t)tok * E_EXP + e] : 0.f;
  }
  __syncthreads();

  // stage x tile -> LDS bf16, dword-packed (coalesced float2 reads, b32 LDS writes)
  {
    unsigned int* xsw = (unsigned int*)xs;
    for (int idx = tid; idx < 16 * (H_DIM / 2); idx += 256) {
      int m = idx >> 9;                 // H_DIM/2 == 512
      int k2 = idx & (H_DIM / 2 - 1);
      float f0 = 0.f, f1 = 0.f;
      if (m < nvalid) {
        const float2 v = *(const float2*)(x + (size_t)toks[m] * H_DIM + 2 * k2);
        f0 = v.x; f1 = v.y;
      }
      xsw[idx] = pack2bf_rne(f0, f1);
    }
  }
  __syncthreads();

  const float* G = gate_w + (size_t)e * H_DIM * I_DIM;
  const float* U = up_w   + (size_t)e * H_DIM * I_DIM;
  const float* D = down_w + (size_t)e * I_DIM * H_DIM;

  v8f yacc[8];                                    // this wave's 128 output columns
#pragma unroll
  for (int tt = 0; tt < 8; ++tt) yacc[tt] = (v8f){};

  const int bn = lane & 15;
  const int brow = (lane < 16) ? 0 : 16;

  for (int i0 = 0; i0 < I_DIM; i0 += 256) {
    // --- gate/up for this wave's 32 intermediate columns ---
    v8f g0 = {}, g1 = {}, u0 = {}, u1 = {};
    const int nb = i0 + wave * 32;
#pragma unroll 1
    for (int kb = 0; kb < H_DIM; kb += 32) {
      // unconditional speculative prefetch of next K-step's weight cachelines
      {
        const size_t roff = (size_t)(kb + 32 + brow) * I_DIM;
        __builtin_prefetch(G + roff + nb + bn, 0, 1);
        __builtin_prefetch(G + roff + nb + 16 + bn, 0, 1);
        __builtin_prefetch(U + roff + nb + bn, 0, 1);
        __builtin_prefetch(U + roff + nb + 16 + bn, 0, 1);
      }
      v16bf a   = load_a_lds(xs, H_DIM, kb, lane);
      v16bf bg0 = load_b_global(G, I_DIM, kb, nb, lane);
      v16bf bg1 = load_b_global(G, I_DIM, kb, nb + 16, lane);
      v16bf bu0 = load_b_global(U, I_DIM, kb, nb, lane);
      v16bf bu1 = load_b_global(U, I_DIM, kb, nb + 16, lane);
      g0 = wmma_bf16(a, bg0, g0);
      g1 = wmma_bf16(a, bg1, g1);
      u0 = wmma_bf16(a, bu0, u0);
      u1 = wmma_bf16(a, bu1, u1);
    }
    __syncthreads();   // all waves done reading hs from previous chunk
    {
      // h = silu(g)*u -> LDS bf16 (C-layout: lanes0-15 N=lane M=r, lanes16-31 N=lane-16 M=r+8)
      const int n = lane & 15;
      const int mo = (lane < 16) ? 0 : 8;
#pragma unroll
      for (int r = 0; r < 8; ++r) {
        hs[(mo + r) * 256 + wave * 32 + n]      = f2bf(fast_silu_mul(g0[r], u0[r]));
        hs[(mo + r) * 256 + wave * 32 + 16 + n] = f2bf(fast_silu_mul(g1[r], u1[r]));
      }
    }
    __syncthreads();
    // --- down projection: y(16x128 per wave) += h(16x256) @ D[i0:i0+256, cols] ---
#pragma unroll 1
    for (int kb = 0; kb < 256; kb += 32) {
      __builtin_prefetch(D + (size_t)(i0 + kb + 32 + brow) * H_DIM + wave * 128 + bn, 0, 1);
      v16bf a2 = load_a_lds(hs, 256, kb, lane);
#pragma unroll
      for (int tt = 0; tt < 8; ++tt) {
        v16bf bd = load_b_global(D, H_DIM, i0 + kb, wave * 128 + tt * 16, lane);
        yacc[tt] = wmma_bf16(a2, bd, yacc[tt]);
      }
    }
  }

  // weighted scatter-add into out
  const int n = lane & 15;
  const int mo = (lane < 16) ? 0 : 8;
#pragma unroll
  for (int tt = 0; tt < 8; ++tt) {
    int hcol = wave * 128 + tt * 16 + n;
#pragma unroll
    for (int r = 0; r < 8; ++r) {
      int m = mo + r;
      if (m < nvalid) {
        unsafeAtomicAdd(&out[(size_t)toks[m] * H_DIM + hcol], wts[m] * yacc[tt][r]);
      }
    }
  }
}

extern "C" void kernel_launch(void* const* d_in, const int* in_sizes, int n_in,
                              void* d_out, int out_size, void* d_ws, size_t ws_size,
                              hipStream_t stream) {
  const float* x      = (const float*)d_in[0];
  const float* Wg     = (const float*)d_in[1];
  const float* gate_w = (const float*)d_in[2];
  const float* up_w   = (const float*)d_in[3];
  const float* down_w = (const float*)d_in[4];
  float* out = (float*)d_out;

  char* ws = (char*)d_ws;
  float* combine = (float*)ws;                                            // T*E floats (64 KB)
  int*   cnt     = (int*)(ws + (size_t)T_TOK * E_EXP * sizeof(float));    // 16 ints
  int*   list    = (int*)(ws + (size_t)T_TOK * E_EXP * sizeof(float) + 256); // E*T ints (64 KB)

  {
    int total = T_TOK * H_DIM;
    zero_kernel<<<(total + 255) / 256, 256, 0, stream>>>(out, cnt);
  }
  router_kernel<<<T_TOK / 4, 128, 0, stream>>>(x, Wg, combine);
  scatter_kernel<<<(T_TOK + 255) / 256, 256, 0, stream>>>(combine, cnt, list);

  dim3 grid(T_TOK / 16, E_EXP);
  moe_ffn_kernel<<<grid, 256, 0, stream>>>(x, gate_w, up_w, down_w, combine, cnt, list, out);
}
